// GraphAttentionLayer_83133386981785
// MI455X (gfx1250) — matrix operand order repaired and checked by reading, not compiled, not run on hardware
//
#include <hip/hip_runtime.h>
#include <hip/hip_bf16.h>
#include <cstdint>

typedef __attribute__((ext_vector_type(16))) _Float16 v16h;
typedef __attribute__((ext_vector_type(8)))  float    v8f;
typedef __attribute__((ext_vector_type(4)))  float    v4f;
typedef __attribute__((ext_vector_type(4)))  int      v4i;

#define GAT_N    8192
#define GAT_IN   256
#define GAT_D    128
#define GAT_SEG  8                     // column segments per 16-row tile
#define GAT_JSEG (GAT_N / GAT_SEG)     // 1024 columns per wave
#define GAT_TILES (GAT_N / 16)         // 512 row tiles
#define GAT_GW   (GAT_TILES * GAT_SEG) // 4096 waves total

// ---------------------------------------------------------------------------
// K1: h = x@W (per-row), write hT (f16, [D][N] transposed), s1 = h.a1, s2 = h.a2
// ---------------------------------------------------------------------------
__global__ void __launch_bounds__(128)
gat_h_kernel(const float* __restrict__ x, const float* __restrict__ W,
             const float* __restrict__ a1, const float* __restrict__ a2,
             _Float16* __restrict__ hT, float* __restrict__ s1,
             float* __restrict__ s2) {
  __shared__ float xs[GAT_IN];
  __shared__ float red[128];
  const int i = blockIdx.x;
  const int t = threadIdx.x;                     // 0..127
  xs[t]       = x[(size_t)i * GAT_IN + t];
  xs[t + 128] = x[(size_t)i * GAT_IN + t + 128];
  __syncthreads();
  float acc = 0.f;
#pragma unroll 8
  for (int k = 0; k < GAT_IN; ++k)
    acc = fmaf(xs[k], W[k * GAT_D + t], acc);
  hT[(size_t)t * GAT_N + i] = (_Float16)acc;

  red[t] = acc * a1[t];
  __syncthreads();
  for (int s = 64; s > 0; s >>= 1) { if (t < s) red[t] += red[t + s]; __syncthreads(); }
  if (t == 0) s1[i] = red[0];
  __syncthreads();
  red[t] = acc * a2[t];
  __syncthreads();
  for (int s = 64; s > 0; s >>= 1) { if (t < s) red[t] += red[t + s]; __syncthreads(); }
  if (t == 0) s2[i] = red[0];
}

// ---------------------------------------------------------------------------
// K2: gmax = max_j s2[j]  (global bound instead of per-row softmax max; exact
// after normalization, numerically safe for score spreads < ~80 — this saves
// an entire 256 MB masked-max pass over adj)
// ---------------------------------------------------------------------------
__global__ void __launch_bounds__(256)
gat_gmax_kernel(const float* __restrict__ s2, float* __restrict__ gm) {
  __shared__ float red[256];
  float m = -__builtin_inff();
  for (int j = threadIdx.x; j < GAT_N; j += 256) m = fmaxf(m, s2[j]);
  red[threadIdx.x] = m;
  __syncthreads();
  for (int s = 128; s > 0; s >>= 1) {
    if (threadIdx.x < s) red[threadIdx.x] = fmaxf(red[threadIdx.x], red[threadIdx.x + s]);
    __syncthreads();
  }
  if (threadIdx.x == 0) gm[0] = red[0];
}

// ---------------------------------------------------------------------------
// K3: fused mask -> exp -> P@H (WMMA f16, f32 accum). Single streaming pass
// over adj (NT loads: 256 MB > 192 MB L2, keep reuse set resident). Wave =
// one 16-row tile x one 1024-column segment (4096 waves for latency hiding).
// ---------------------------------------------------------------------------
__global__ void __launch_bounds__(256)
gat_attn_partial(const int* __restrict__ adj, const _Float16* __restrict__ hT,
                 const float* __restrict__ s1, const float* __restrict__ s2,
                 const float* __restrict__ gmp, float* __restrict__ Opart,
                 float* __restrict__ RSpart) {
  const int lane  = threadIdx.x & 31;
  const int wave  = threadIdx.x >> 5;
  const int gw    = blockIdx.x * 8 + wave;       // 0..GAT_GW-1
  const int itile = gw / GAT_SEG;                // row tile
  const int seg   = gw % GAT_SEG;                // column segment
  const int i0    = itile * 16;
  const int mloc  = lane & 15;                   // A-frag row / B-frag col
  const int kg    = lane >> 4;                   // K-group select
  const int row   = i0 + mloc;
  const int jbeg  = seg * GAT_JSEG;

  const float s1v  = s1[row];
  const float gmax = gmp[0];
  const float zs   = s1v + gmax;
  const float smax = zs > 0.f ? zs : 0.01f * zs; // LRelu(s1_i + gmax) >= row max

  v8f acc[8] = {};
  float rowsum = 0.f;

  const int* adjRow = adj + (size_t)row * GAT_N;
  const _Float16* bBase = hT + (size_t)mloc * GAT_N + kg * 16;

  for (int j = jbeg; j < jbeg + GAT_JSEG; j += 32) {
    // adj tile 16x32 in f16 A-fragment lane mapping:
    // lane l: row = l&15; K = (l>>4)*16 + 0..15
    const int c1 = j + kg * 16;
    const int c2 = c1 + 8;
    const v4i A0 = __builtin_nontemporal_load((const v4i*)(adjRow + c1));
    const v4i A1 = __builtin_nontemporal_load((const v4i*)(adjRow + c1 + 4));
    const v4i A2 = __builtin_nontemporal_load((const v4i*)(adjRow + c2));
    const v4i A3 = __builtin_nontemporal_load((const v4i*)(adjRow + c2 + 4));
    const v4f S0 = *(const v4f*)(s2 + c1);
    const v4f S1 = *(const v4f*)(s2 + c1 + 4);
    const v4f S2 = *(const v4f*)(s2 + c2);
    const v4f S3 = *(const v4f*)(s2 + c2 + 4);
    const int   av[16] = {A0[0], A0[1], A0[2], A0[3], A1[0], A1[1], A1[2], A1[3],
                          A2[0], A2[1], A2[2], A2[3], A3[0], A3[1], A3[2], A3[3]};
    const float sv[16] = {S0[0], S0[1], S0[2], S0[3], S1[0], S1[1], S1[2], S1[3],
                          S2[0], S2[1], S2[2], S2[3], S3[0], S3[1], S3[2], S3[3]};
    v16h aF;
#pragma unroll
    for (int e = 0; e < 16; ++e) {
      float z = s1v + sv[e];
      z = z > 0.f ? z : 0.01f * z;               // leaky relu
      float p = (av[e] != 0) ? __expf(z - smax) : 0.0f;
      rowsum += p;
      aF[e] = (_Float16)p;
    }
    // 8 N-tiles of 16 columns: B-frag lane l: col = nt*16 + (l&15),
    // K rows j + (l>>4)*16 + 0..15 -> contiguous 32B from transposed hT.
#pragma unroll
    for (int nt = 0; nt < 8; ++nt) {
      const v16h bF = *(const v16h*)(bBase + (size_t)nt * 16 * GAT_N + j);
      acc[nt] = __builtin_amdgcn_wmma_f32_16x16x32_f16(
          false, aF, false, bF, (short)0, acc[nt], false, false);
    }
  }

  // lanes l and l^16 hold the two K-halves of row (l&15)
  const float rsAll = rowsum + __shfl_xor(rowsum, 16, 32);
  if (lane < 16) RSpart[gw * 16 + lane] = rsAll;

  // C/D layout: VGPR r, lane l -> row r + (l>>4)*8, col l&15
  float* Obase = Opart + (size_t)gw * (16 * GAT_D);
#pragma unroll
  for (int r = 0; r < 8; ++r) {
    const int rr = kg * 8 + r;
#pragma unroll
    for (int nt = 0; nt < 8; ++nt)
      __builtin_nontemporal_store(acc[nt][r], Obase + rr * GAT_D + nt * 16 + mloc);
  }
}

// ---------------------------------------------------------------------------
// K4: combine segment partials, normalize (empty rows -> 0, matches
// nan_to_num(softmax) semantics), apply ELU, write f32 output.
// ---------------------------------------------------------------------------
__global__ void __launch_bounds__(256)
gat_finalize(const float* __restrict__ Opart, const float* __restrict__ RSpart,
             float* __restrict__ out) {
  const int idx = blockIdx.x * 256 + threadIdx.x;   // 0..N*D-1
  const int i  = idx >> 7;
  const int c  = idx & 127;
  const int itile = i >> 4;
  const int rr    = i & 15;
  float o = 0.f, rs = 0.f;
#pragma unroll
  for (int g = 0; g < GAT_SEG; ++g) {
    const int gw = itile * GAT_SEG + g;
    o  += __builtin_nontemporal_load(Opart + (size_t)gw * (16 * GAT_D) + rr * GAT_D + c);
    rs += RSpart[gw * 16 + rr];
  }
  const float inv = rs > 0.f ? 1.0f / rs : 0.f;
  const float v = o * inv;
  out[idx] = v > 0.f ? v : __expf(v) - 1.0f;        // ELU(alpha=1)
}

// ---------------------------------------------------------------------------
extern "C" void kernel_launch(void* const* d_in, const int* in_sizes, int n_in,
                              void* d_out, int out_size, void* d_ws, size_t ws_size,
                              hipStream_t stream) {
  const float* x   = (const float*)d_in[0];
  const int*   adj = (const int*)d_in[1];
  const float* W   = (const float*)d_in[2];
  const float* a1  = (const float*)d_in[3];
  const float* a2  = (const float*)d_in[4];
  float* out = (float*)d_out;

  char* ws = (char*)d_ws;
  _Float16* hT = (_Float16*)ws;                       // 128*8192*2 = 2 MB
  float* s1 = (float*)(ws + (size_t)2 * 1024 * 1024); // 8192 f32
  float* s2 = s1 + GAT_N;                             // 8192 f32
  float* gm = s2 + GAT_N;                             // 1 f32 (padded)
  float* RSpart = gm + 64;                            // 4096*16 f32 = 256 KB
  float* Opart  = RSpart + (size_t)GAT_GW * 16;       // 4096*2048 f32 = 32 MB

  gat_h_kernel<<<GAT_N, 128, 0, stream>>>(x, W, a1, a2, hT, s1, s2);
  gat_gmax_kernel<<<1, 256, 0, stream>>>(s2, gm);
  gat_attn_partial<<<GAT_GW / 8, 256, 0, stream>>>(adj, hT, s1, s2, gm, Opart, RSpart);
  gat_finalize<<<(GAT_N * GAT_D) / 256, 256, 0, stream>>>(Opart, RSpart, out);
}
